// PositiveMultiheadAttention_7000796692902
// MI455X (gfx1250) — matrix operand order repaired and checked by reading, not compiled, not run on hardware
//
#include <hip/hip_runtime.h>
#include <hip/hip_bf16.h>
#include <math.h>
#include <stdint.h>

// ---- problem constants (from setup_inputs) ----
#define BB 4
#define LL 4096
#define DD 1024
#define HH 16
#define DH 64
#define MT (BB * LL)   // 16384 token rows

typedef __attribute__((ext_vector_type(16))) __bf16 v16bf;
typedef __attribute__((ext_vector_type(8)))  __bf16 v8bf;
typedef __attribute__((ext_vector_type(8)))  float  v8f;
typedef __attribute__((ext_vector_type(4)))  int    v4i;

#define DEV __device__ __forceinline__

// ---- CDNA5 async global->LDS staging (ASYNCcnt-tracked DMA, no VGPR hop) ----
#if defined(__has_builtin)
#  if __has_builtin(__builtin_amdgcn_global_load_async_to_lds_b128)
#    define USE_ASYNC_LDS 1
#  endif
#  if __has_builtin(__builtin_amdgcn_s_wait_asynccnt)
#    define HAVE_WAIT_ASYNC_BUILTIN 1
#  endif
#endif
#ifndef USE_ASYNC_LDS
#  define USE_ASYNC_LDS 0
#endif

// Builtin signature (from clang diagnostic): (v4i AS1*, v4i AS3*, imm int, imm int)
#define GLBP(p) ((__attribute__((address_space(1))) v4i*)(uintptr_t)(p))
#define LDSP(p) ((__attribute__((address_space(3))) v4i*)(unsigned)(uintptr_t)(p))

// Copy 16B (8 x bf16) global -> LDS for this lane.
static DEV void stage16(__bf16* l, const __bf16* g) {
#if USE_ASYNC_LDS
  __builtin_amdgcn_global_load_async_to_lds_b128(GLBP(g), LDSP(l), 0, 0);
#else
  *(v8bf*)l = *(const v8bf*)g;
#endif
}

static DEV void stage16_wait() {
#if USE_ASYNC_LDS
#  if defined(HAVE_WAIT_ASYNC_BUILTIN)
  __builtin_amdgcn_s_wait_asynccnt(0);
#  else
  asm volatile("s_wait_asynccnt 0x0" ::: "memory");
#  endif
#endif
}

static DEV v8f wmma_bf16(v16bf a, v16bf b, v8f c) {
  // D = A(16x32 bf16) * B(32x16 bf16) + C(16x16 f32)
  return __builtin_amdgcn_wmma_f32_16x16x32_bf16(false, a, false, b, (short)0, c,
                                                 false, false);
}

// A-fragment gather from a row-major [16 x >=32] bf16 tile with row stride ld.
// Lane l (wave32): row m = l&15; half h = l>>4; K = {h*8..h*8+7, 16+h*8..16+h*8+7}.
static DEV v16bf load_frag_a(const __bf16* t, int ld) {
  const int lane = threadIdx.x & 31;
  const int r = lane & 15, hf = lane >> 4;
  const __bf16* p = t + r * ld;
  v8bf lo = *(const v8bf*)(p + hf * 8);
  v8bf hi = *(const v8bf*)(p + 16 + hf * 8);
  v16bf f;
#pragma unroll
  for (int i = 0; i < 8; ++i) { f[i] = lo[i]; f[8 + i] = hi[i]; }
  return f;
}

// B-fragment gather from a transposed tile Bt[n][k] (row-major, stride ld).
// Lane l: col n = l&15; half h = l>>4; K = {h*8..h*8+7, 16+h*8..16+h*8+7}.
static DEV v16bf load_frag_b(const __bf16* t, int ld) {
  const int lane = threadIdx.x & 31;
  const int r = lane & 15, hf = lane >> 4;
  const __bf16* p = t + r * ld;
  v8bf lo = *(const v8bf*)(p + hf * 8);
  v8bf hi = *(const v8bf*)(p + 16 + hf * 8);
  v16bf f;
#pragma unroll
  for (int i = 0; i < 8; ++i) { f[i] = lo[i]; f[8 + i] = hi[i]; }
  return f;
}

// Sum across the 16-lane group sharing bit4 (C/D rows live in one half-wave).
static DEV float rowsum16(float x) {
  x += __shfl_xor(x, 1, 32);
  x += __shfl_xor(x, 2, 32);
  x += __shfl_xor(x, 4, 32);
  x += __shfl_xor(x, 8, 32);
  return x;
}

static DEV float sigmoidf(float x) { return 1.0f / (1.0f + __expf(-x)); }

// ---------------------------------------------------------------------------
// Kernel 1: fused QKV projection  C[m][n] = sum_d X[m][d] * W[n][d] + bias
// z=0: Q -> sigmoid -> row-l2norm (per 64-wide head chunk) -> bf16 head-major
// z=1: K -> sigmoid -> row-l2norm (1/L2 scale cancels under l2norm) -> bf16
// z=2: V -> bias, split max/min with key mask folded in -> Vp / Vm bf16
// Tile: 128(M) x 128(N) x 32(K); 8 waves in a 4x2 grid; each wave 32x64.
// (f32 sources need conversion, so staging stays on the VALU path here.)
// ---------------------------------------------------------------------------
__global__ __launch_bounds__(256) void proj_qkv_kernel(
    const float* __restrict__ Xq, const float* __restrict__ Xk, const float* __restrict__ Xv,
    const float* __restrict__ Wq, const float* __restrict__ Wk, const float* __restrict__ Wv,
    const float* __restrict__ bq, const float* __restrict__ bk, const float* __restrict__ bv,
    const float* __restrict__ mask,
    __bf16* __restrict__ Qh, __bf16* __restrict__ Kh,
    __bf16* __restrict__ Vp, __bf16* __restrict__ Vm) {
  __shared__ __align__(16) __bf16 At[128 * 32];
  __shared__ __align__(16) __bf16 Bt[128 * 32];

  const int z = blockIdx.z;
  const float* X    = (z == 0) ? Xq : (z == 1) ? Xk : Xv;
  const float* W    = (z == 0) ? Wq : (z == 1) ? Wk : Wv;
  const float* bias = (z == 0) ? bq : (z == 1) ? bk : bv;

  const int m0 = blockIdx.x * 128;
  const int n0 = blockIdx.y * 128;
  const int t = threadIdx.x;
  const int wave = t >> 5, lane = t & 31;
  const int wm = wave & 3, wn = wave >> 2;

  v8f acc[2][4] = {};

  for (int k0 = 0; k0 < DD; k0 += 32) {
    __syncthreads();
#pragma unroll
    for (int j = 0; j < 2; ++j) {
      const int chunk = t + j * 256;         // 512 chunks of 8 elements
      const int r = chunk >> 2;              // tile row 0..127
      const int c8 = (chunk & 3) * 8;        // k sub-offset
      const float* sa = X + (size_t)(m0 + r) * DD + k0 + c8;
      const float* sb = W + (size_t)(n0 + r) * DD + k0 + c8;  // Bt[n][k] = W[n][k]
      if (k0 + 32 < DD) {
        __builtin_prefetch(sa + 32, 0, 1);   // global_prefetch_b8 next K tile
        __builtin_prefetch(sb + 32, 0, 1);
      }
      v8bf va, vb;
#pragma unroll
      for (int e = 0; e < 8; ++e) { va[e] = (__bf16)sa[e]; vb[e] = (__bf16)sb[e]; }
      *(v8bf*)(At + r * 32 + c8) = va;
      *(v8bf*)(Bt + r * 32 + c8) = vb;
    }
    __syncthreads();

    v16bf a[2], b[4];
#pragma unroll
    for (int i = 0; i < 2; ++i) a[i] = load_frag_a(At + (wm * 32 + i * 16) * 32, 32);
#pragma unroll
    for (int j = 0; j < 4; ++j) b[j] = load_frag_b(Bt + (wn * 64 + j * 16) * 32, 32);
#pragma unroll
    for (int i = 0; i < 2; ++i)
#pragma unroll
      for (int j = 0; j < 4; ++j) acc[i][j] = wmma_bf16(a[i], b[j], acc[i][j]);
  }

  const int hf = lane >> 4, lc = lane & 15;
#pragma unroll
  for (int i = 0; i < 2; ++i) {
    const int mbase = m0 + wm * 32 + i * 16 + hf * 8;  // + r gives global row
    if (z < 2) {
      // sigmoid + per-row L2 norm over the 64-wide head chunk this wave owns
      float val[4][8], ss[8];
#pragma unroll
      for (int r = 0; r < 8; ++r) ss[r] = 0.f;
#pragma unroll
      for (int j = 0; j < 4; ++j) {
        const int n = n0 + wn * 64 + j * 16 + lc;
        const float bn = bias[n];
#pragma unroll
        for (int r = 0; r < 8; ++r) {
          const float v = sigmoidf(acc[i][j][r] + bn);
          val[j][r] = v;
          ss[r] += v * v;
        }
      }
      float sc[8];
#pragma unroll
      for (int r = 0; r < 8; ++r)
        sc[r] = 1.0f / fmaxf(sqrtf(rowsum16(ss[r])), 1e-12f);
      __bf16* dst = (z == 0) ? Qh : Kh;
#pragma unroll
      for (int j = 0; j < 4; ++j) {
        const int n = n0 + wn * 64 + j * 16 + lc;
        const int h = n >> 6, c = n & 63;
#pragma unroll
        for (int r = 0; r < 8; ++r) {
          const int m = mbase + r;
          const int b = m >> 12, l = m & (LL - 1);
          dst[(((size_t)(b * HH + h)) * LL + l) * DH + c] = (__bf16)(val[j][r] * sc[r]);
        }
      }
    } else {
      // V: bias, then masked positive/negative split (mask folds into kv einsum)
#pragma unroll
      for (int j = 0; j < 4; ++j) {
        const int n = n0 + wn * 64 + j * 16 + lc;
        const int h = n >> 6, c = n & 63;
        const float bn = bias[n];
#pragma unroll
        for (int r = 0; r < 8; ++r) {
          const int m = mbase + r;
          const int b = m >> 12, l = m & (LL - 1);
          const float mk = mask[(size_t)b * LL + l];
          const float v = acc[i][j][r] + bn;
          const size_t idx = (((size_t)(b * HH + h)) * LL + l) * DH + c;
          Vp[idx] = (__bf16)(fmaxf(v, 0.f) * mk);
          Vm[idx] = (__bf16)(fminf(v, 0.f) * mk);
        }
      }
    }
  }
}

// ---------------------------------------------------------------------------
// Kernel 2: kv[b,h,d,c] = sum_l K[l,d] * Vsign[l,c]   (64x64, K-loop = 4096)
// One block per (b*h, sign); 8 waves: wave w -> mSub = w>>1, two nSubs.
// Both operands are l-major in memory, so stage transposed tiles in LDS
// (2-byte scatter transpose cannot ride the 16B async-DMA path).
// ---------------------------------------------------------------------------
__global__ __launch_bounds__(256) void kv_kernel(
    const __bf16* __restrict__ Kh, const __bf16* __restrict__ Vp,
    const __bf16* __restrict__ Vm, float* __restrict__ kvp,
    float* __restrict__ kvm) {
  __shared__ __align__(16) __bf16 At[64 * 32];  // [d][l]
  __shared__ __align__(16) __bf16 Bt[64 * 32];  // [c][l]

  const int bh = blockIdx.x;
  const int s = blockIdx.y;
  const __bf16* V = s ? Vm : Vp;
  float* out = s ? kvm : kvp;
  const __bf16* Kbase = Kh + (size_t)bh * LL * DH;
  const __bf16* Vbase = V + (size_t)bh * LL * DH;

  const int t = threadIdx.x;
  const int wave = t >> 5, lane = t & 31;
  const int wm = wave >> 1;            // d subtile 0..3
  const int wnb = (wave & 1) * 2;      // c subtile base {0,2}

  v8f acc[2] = {};

  for (int l0 = 0; l0 < LL; l0 += 32) {
    __syncthreads();
#pragma unroll
    for (int j = 0; j < 8; ++j) {       // 2048 elems per tile / 256 threads
      const int e = t + j * 256;
      const int li = e >> 6;            // 0..31
      const int xi = e & 63;            // 0..63
      At[xi * 32 + li] = Kbase[(size_t)(l0 + li) * DH + xi];
      Bt[xi * 32 + li] = Vbase[(size_t)(l0 + li) * DH + xi];
    }
    __syncthreads();
    const v16bf a = load_frag_a(At + (wm * 16) * 32, 32);
#pragma unroll
    for (int j = 0; j < 2; ++j) {
      const v16bf b = load_frag_b(Bt + ((wnb + j) * 16) * 32, 32);
      acc[j] = wmma_bf16(a, b, acc[j]);
    }
  }

  const int hf = lane >> 4, lc = lane & 15;
#pragma unroll
  for (int j = 0; j < 2; ++j)
#pragma unroll
    for (int r = 0; r < 8; ++r) {
      const int d = wm * 16 + hf * 8 + r;
      const int c = (wnb + j) * 16 + lc;
      out[(size_t)bh * (DH * DH) + d * DH + c] = acc[j][r];
    }
}

// ---------------------------------------------------------------------------
// Kernel 3: heads = l2norm(Q @ kv_plus) + l2norm(Q @ kv_minus)  -> bf16 (b,l,D)
// Block: 128 l-rows x 64 c-cols, K = 64. Plus/minus share A-fragments.
// Q tile is a pure bf16 copy -> async global->LDS DMA; kv^T staging (f32->bf16
// + transpose) runs on the VALU path and overlaps with the async transfers.
// ---------------------------------------------------------------------------
__global__ __launch_bounds__(256) void heads_kernel(
    const __bf16* __restrict__ Qh, const float* __restrict__ kvp,
    const float* __restrict__ kvm, __bf16* __restrict__ hs) {
  __shared__ __align__(16) __bf16 At[128 * 64];  // Q rows [l][d]
  __shared__ __align__(16) __bf16 Bp[64 * 64];   // kv_plus^T  [c][d]
  __shared__ __align__(16) __bf16 Bm[64 * 64];   // kv_minus^T [c][d]

  const int l0 = blockIdx.x * 128;
  const int bh = blockIdx.y;
  const int b = bh >> 4, h = bh & 15;
  const int t = threadIdx.x;
  const int wave = t >> 5, lane = t & 31;

  // issue the async Q-tile DMA first so it overlaps the kv^T conversion below
  const __bf16* Qbase = Qh + ((size_t)bh * LL + l0) * DH;
#pragma unroll
  for (int j = 0; j < 4; ++j) {        // 8192 elems in 8-wide chunks
    const int chunk = t + j * 256;
    const int r = chunk >> 3;
    const int c8 = (chunk & 7) * 8;
    stage16(At + r * 64 + c8, Qbase + (size_t)r * DH + c8);
  }

  const float* kp = kvp + (size_t)bh * (DH * DH);
  const float* km = kvm + (size_t)bh * (DH * DH);
#pragma unroll
  for (int j = 0; j < 16; ++j) {       // 4096 elems / 256 threads
    const int e = t + j * 256;
    const int d = e >> 6, c = e & 63;
    Bp[c * 64 + d] = (__bf16)kp[e];
    Bm[c * 64 + d] = (__bf16)km[e];
  }
  stage16_wait();
  __syncthreads();

  v8f ap[4] = {}, am[4] = {};
#pragma unroll
  for (int k = 0; k < 2; ++k) {
    const v16bf a = load_frag_a(At + (wave * 16) * 64 + k * 32, 64);
#pragma unroll
    for (int j = 0; j < 4; ++j) {
      const v16bf bp = load_frag_b(Bp + (j * 16) * 64 + k * 32, 64);
      const v16bf bm = load_frag_b(Bm + (j * 16) * 64 + k * 32, 64);
      ap[j] = wmma_bf16(a, bp, ap[j]);
      am[j] = wmma_bf16(a, bm, am[j]);
    }
  }

  const int hf = lane >> 4, lc = lane & 15;
  float ssp[8], ssm[8];
#pragma unroll
  for (int r = 0; r < 8; ++r) { ssp[r] = 0.f; ssm[r] = 0.f; }
#pragma unroll
  for (int j = 0; j < 4; ++j)
#pragma unroll
    for (int r = 0; r < 8; ++r) {
      ssp[r] += ap[j][r] * ap[j][r];
      ssm[r] += am[j][r] * am[j][r];
    }
  float sp[8], sm[8];
#pragma unroll
  for (int r = 0; r < 8; ++r) {
    sp[r] = 1.0f / fmaxf(sqrtf(rowsum16(ssp[r])), 1e-12f);
    sm[r] = 1.0f / fmaxf(sqrtf(rowsum16(ssm[r])), 1e-12f);
  }
#pragma unroll
  for (int j = 0; j < 4; ++j)
#pragma unroll
    for (int r = 0; r < 8; ++r) {
      const int c = j * 16 + lc;
      const int l = l0 + wave * 16 + hf * 8 + r;
      const float v = ap[j][r] * sp[r] + am[j][r] * sm[r];
      hs[((size_t)b * LL + l) * DD + h * DH + c] = (__bf16)v;
    }
}

// ---------------------------------------------------------------------------
// Kernel 4: out[m][n] = sum_c hs[m][c] * Wo[n][c] + bo[n]   (f32 result)
// A-tile (bf16 hs) staged by async global->LDS DMA; Wo tile converts f32->bf16
// on the VALU while the DMA is in flight.
// ---------------------------------------------------------------------------
__global__ __launch_bounds__(256) void out_kernel(
    const __bf16* __restrict__ hs, const float* __restrict__ Wo,
    const float* __restrict__ bo, float* __restrict__ out) {
  __shared__ __align__(16) __bf16 At[128 * 32];
  __shared__ __align__(16) __bf16 Bt[128 * 32];

  const int m0 = blockIdx.x * 128;
  const int n0 = blockIdx.y * 128;
  const int t = threadIdx.x;
  const int wave = t >> 5, lane = t & 31;
  const int wm = wave & 3, wn = wave >> 2;

  v8f acc[2][4] = {};

  for (int k0 = 0; k0 < DD; k0 += 32) {
    __syncthreads();
    // 1) kick off async A-tile DMA
#pragma unroll
    for (int j = 0; j < 2; ++j) {
      const int chunk = t + j * 256;
      const int r = chunk >> 2;
      const int c8 = (chunk & 3) * 8;
      stage16(At + r * 32 + c8, hs + (size_t)(m0 + r) * DD + k0 + c8);
    }
    // 2) convert Wo f32 -> bf16 into LDS while the DMA runs
#pragma unroll
    for (int j = 0; j < 2; ++j) {
      const int chunk = t + j * 256;
      const int r = chunk >> 2;
      const int c8 = (chunk & 3) * 8;
      const float* sb = Wo + (size_t)(n0 + r) * DD + k0 + c8;
      if (k0 + 32 < DD) __builtin_prefetch(sb + 32, 0, 1);
      v8bf vb;
#pragma unroll
      for (int e = 0; e < 8; ++e) vb[e] = (__bf16)sb[e];
      *(v8bf*)(Bt + r * 32 + c8) = vb;
    }
    stage16_wait();
    __syncthreads();

    v16bf a[2], b[4];
#pragma unroll
    for (int i = 0; i < 2; ++i) a[i] = load_frag_a(At + (wm * 32 + i * 16) * 32, 32);
#pragma unroll
    for (int j = 0; j < 4; ++j) b[j] = load_frag_b(Bt + (wn * 64 + j * 16) * 32, 32);
#pragma unroll
    for (int i = 0; i < 2; ++i)
#pragma unroll
      for (int j = 0; j < 4; ++j) acc[i][j] = wmma_bf16(a[i], b[j], acc[i][j]);
  }

  const int hf = lane >> 4, lc = lane & 15;
#pragma unroll
  for (int i = 0; i < 2; ++i)
#pragma unroll
    for (int j = 0; j < 4; ++j) {
      const int n = n0 + wn * 64 + j * 16 + lc;
      const float bn = bo[n];
#pragma unroll
      for (int r = 0; r < 8; ++r) {
        const int m = m0 + wm * 32 + i * 16 + hf * 8 + r;
        out[(size_t)m * DD + n] = acc[i][j][r] + bn;
      }
    }
}

// ---------------------------------------------------------------------------
extern "C" void kernel_launch(void* const* d_in, const int* in_sizes, int n_in,
                              void* d_out, int out_size, void* d_ws, size_t ws_size,
                              hipStream_t stream) {
  (void)in_sizes; (void)n_in; (void)out_size; (void)ws_size;
  // setup_inputs order: queries, values, keys, key_attention_mask,
  //                     Wq, bq, Wk, bk, Wv, bv, Wo, bo
  const float* queries = (const float*)d_in[0];
  const float* values  = (const float*)d_in[1];
  const float* keys    = (const float*)d_in[2];
  const float* mask    = (const float*)d_in[3];
  const float* Wq = (const float*)d_in[4];
  const float* bq = (const float*)d_in[5];
  const float* Wk = (const float*)d_in[6];
  const float* bk = (const float*)d_in[7];
  const float* Wv = (const float*)d_in[8];
  const float* bv = (const float*)d_in[9];
  const float* Wo = (const float*)d_in[10];
  const float* bo = (const float*)d_in[11];

  // workspace carve-out (~170 MB total)
  char* ws = (char*)d_ws;
  const size_t szHead = (size_t)BB * HH * LL * DH * 2;  // bf16 head-major tensor
  __bf16* Qh = (__bf16*)ws; ws += szHead;
  __bf16* Kh = (__bf16*)ws; ws += szHead;
  __bf16* Vp = (__bf16*)ws; ws += szHead;
  __bf16* Vm = (__bf16*)ws; ws += szHead;
  float* kvp = (float*)ws;  ws += (size_t)BB * HH * DH * DH * 4;
  float* kvm = (float*)ws;  ws += (size_t)BB * HH * DH * DH * 4;
  __bf16* hs = (__bf16*)ws; ws += (size_t)MT * DD * 2;

  proj_qkv_kernel<<<dim3(MT / 128, DD / 128, 3), 256, 0, stream>>>(
      queries, keys, values, Wq, Wk, Wv, bq, bk, bv, mask, Qh, Kh, Vp, Vm);
  kv_kernel<<<dim3(BB * HH, 2), 256, 0, stream>>>(Kh, Vp, Vm, kvp, kvm);
  heads_kernel<<<dim3(LL / 128, BB * HH), 256, 0, stream>>>(Qh, kvp, kvm, hs);
  out_kernel<<<dim3(MT / 128, DD / 128), 256, 0, stream>>>(hs, Wo, bo, (float*)d_out);
}
